// VariationalGCNMLPEncoder_89618787598756
// MI455X (gfx1250) — matrix operand order repaired and checked by reading, not compiled, not run on hardware
//
#include <hip/hip_runtime.h>

// ---------------------------------------------------------------------------
// Types for CDNA5 WMMA (gfx1250, wave32)
// ---------------------------------------------------------------------------
typedef __attribute__((ext_vector_type(16))) __bf16 v16bf;
typedef __attribute__((ext_vector_type(8)))  float  v8f;

union FragBF {
    v16bf v;
    uint4 q[2];   // 2 x 16B = 16 bf16
};

// ---------------------------------------------------------------------------
// bf16 <-> f32 helpers (round-to-nearest-even for f32->bf16)
// ---------------------------------------------------------------------------
__device__ __forceinline__ float bfraw2f(unsigned raw16) {
    union { unsigned u; float f; } v; v.u = raw16 << 16; return v.f;
}
__device__ __forceinline__ unsigned short f2bfraw(float f) {
    union { float f; unsigned u; } v; v.f = f;
    unsigned r = v.u + 0x7FFFu + ((v.u >> 16) & 1u);
    return (unsigned short)(r >> 16);
}
__device__ __forceinline__ __bf16 f2bf(float f) {
    union { unsigned short s; __bf16 b; } o; o.s = f2bfraw(f); return o.b;
}
__device__ __forceinline__ float bf2f(__bf16 b) {
    union { __bf16 b; unsigned short s; } u; u.b = b; return bfraw2f(u.s);
}
__device__ __forceinline__ float lrelu(float t) { return t >= 0.f ? t : 0.01f * t; }

// ---------------------------------------------------------------------------
// Elementwise / prep kernels
// ---------------------------------------------------------------------------
__global__ void k_f32_to_bf16(const float* __restrict__ in, __bf16* __restrict__ out, long n) {
    long i = (long)blockIdx.x * blockDim.x + threadIdx.x;
    if (i < n) out[i] = f2bf(in[i]);
}

// W [K, Nf] f32 row-major  ->  Wt [Nf, K] bf16 row-major (i.e. B columns contiguous)
__global__ void k_transpose_bf16(const float* __restrict__ W, __bf16* __restrict__ Wt,
                                 int K, int Nf) {
    int idx = blockIdx.x * blockDim.x + threadIdx.x;
    if (idx >= K * Nf) return;
    int k = idx / Nf, n = idx - k * Nf;
    Wt[(size_t)n * K + k] = f2bf(W[idx]);
}

__global__ void k_fill_f32(float* __restrict__ p, float val, long n) {
    long i = (long)blockIdx.x * blockDim.x + threadIdx.x;
    if (i < n) p[i] = val;
}

// degree of (A + I): deg initialized to 1.0, then +1 per incoming edge
__global__ void k_deg_count(const long long* __restrict__ ei, float* __restrict__ deg, long E) {
    long e = (long)blockIdx.x * blockDim.x + threadIdx.x;
    if (e < E) atomicAdd(&deg[(int)ei[E + e]], 1.0f);
}

__global__ void k_rsqrt(const float* __restrict__ deg, float* __restrict__ dis, int n) {
    int i = blockIdx.x * blockDim.x + threadIdx.x;
    if (i < n) dis[i] = rsqrtf(deg[i]);
}

// ---------------------------------------------------------------------------
// WMMA GEMM:  Out[M,Nf] = A[M,K](bf16) * Bt[Nf,K](bf16)^T  (+bias)(+leakyrelu)
// Each wave computes a 16x64 output strip (4 adjacent 16x16 N-tiles): the A
// fragment is loaded once per K-step and reused across 4 WMMAs (4x less A
// traffic, 2.5 loads/WMMA instead of 4). 8 waves / 256-thread block.
// mode 0: store bf16, no bias / no act      (conv pre-aggregation h)
// mode 1: store bf16, +bias, leaky-relu     (MLP hidden)
// mode 2: store f32,  +bias, no act         (MLP output -> d_out)
// Fragment loads follow the CDNA5 ISA VGPR layouts (CDNA5 ISA §7.12.2):
//   A (16x32 bf16): lane<16 holds K {0..7,16..23}, lane>=16 holds K {8..15,24..31}
//   B (32x16 bf16): lane<16 holds K {0..7,16..23}, lane>=16 holds K {8..15,24..31}
// ---------------------------------------------------------------------------
__global__ void k_gemm_wmma(const __bf16* __restrict__ A, const __bf16* __restrict__ Bt,
                            const float* __restrict__ bias, void* __restrict__ out,
                            int M, int K, int Nf, int mode) {
    const int lane = threadIdx.x & 31;
    const int wave = threadIdx.x >> 5;
    const int m0 = (blockIdx.x * 8 + wave) * 16;
    const int n0 = blockIdx.y * 64;            // 4 N-tiles per wave
    if (m0 >= M) return;                       // wave-uniform; EXEC all-ones for WMMA below

    const int r  = lane & 15;
    const int hi = lane >> 4;
    const __bf16* Ar = A + (size_t)(m0 + r) * K;     // row m0+r of A
    const __bf16* Br[4];
#pragma unroll
    for (int j = 0; j < 4; ++j)                      // column n0+16j+r of B
        Br[j] = Bt + (size_t)(n0 + 16 * j + r) * K;
    const int kaoff = hi * 8;                        // A: 8-elem chunks at +0/+8 and +16
    const int kboff = hi * 8;                        // B: 8-elem chunks at +0/+8 and +16

    v8f acc[4] = {};
    for (int k = 0; k < K; k += 32) {
        FragBF a;
        a.q[0] = *(const uint4*)(Ar + k + kaoff);
        a.q[1] = *(const uint4*)(Ar + k + kaoff + 16);
#pragma unroll
        for (int j = 0; j < 4; ++j) {
            FragBF b;
            b.q[0] = *(const uint4*)(Br[j] + k + kboff);
            b.q[1] = *(const uint4*)(Br[j] + k + kboff + 16);
            acc[j] = __builtin_amdgcn_wmma_f32_16x16x32_bf16(
                         false, a.v, false, b.v, (short)0, acc[j], false, false);
        }
    }

    // C/D layout: VGPR v -> row m0 + v + (lane>=16 ? 8 : 0), col ntile + (lane&15)
    const int rowbase = m0 + hi * 8;
#pragma unroll
    for (int j = 0; j < 4; ++j) {
        const int col = n0 + 16 * j + r;
        if (mode == 0) {
            __bf16* O = (__bf16*)out;
#pragma unroll
            for (int v = 0; v < 8; ++v)
                O[(size_t)(rowbase + v) * Nf + col] = f2bf(acc[j][v]);
        } else if (mode == 1) {
            __bf16* O = (__bf16*)out;
            const float bv = bias[col];
#pragma unroll
            for (int v = 0; v < 8; ++v)
                O[(size_t)(rowbase + v) * Nf + col] = f2bf(lrelu(acc[j][v] + bv));
        } else {
            float* O = (float*)out;
            const float bv = bias[col];
#pragma unroll
            for (int v = 0; v < 8; ++v)
                O[(size_t)(rowbase + v) * Nf + col] = acc[j][v] + bv;
        }
    }
}

// ---------------------------------------------------------------------------
// Edge scatter: agg[dst,:] += h[src,:] * dis[src]*dis[dst]
// One edge per wave (8 edges / 256-thread block); each lane owns 4 features.
// h is bf16 -> gather traffic is halved vs fp32 (the HBM-dominant term).
// ---------------------------------------------------------------------------
__global__ void k_scatter(const long long* __restrict__ ei, const __bf16* __restrict__ h,
                          const float* __restrict__ dis, float* __restrict__ agg, long E) {
    const int lane = threadIdx.x & 31;
    const long e = (long)blockIdx.x * 8 + (threadIdx.x >> 5);
    if (e >= E) return;
    const int src = (int)ei[e];
    const int dst = (int)ei[E + e];
    const float norm = dis[src] * dis[dst];
    const unsigned* hp = (const unsigned*)(h + (size_t)src * 128);
    const uint2 d = *(const uint2*)(hp + lane * 2);      // 4 bf16 = 8 bytes
    float* ap = agg + (size_t)dst * 128 + lane * 4;
    atomicAdd(ap + 0, bfraw2f(d.x & 0xFFFFu) * norm);
    atomicAdd(ap + 1, bfraw2f(d.x >> 16)     * norm);
    atomicAdd(ap + 2, bfraw2f(d.y & 0xFFFFu) * norm);
    atomicAdd(ap + 3, bfraw2f(d.y >> 16)     * norm);
}

// out = bf16( leaky_relu( agg + h_pre * dis^2 + bias ) )   [self-loop + bias + act]
__global__ void k_finalize(const float* __restrict__ agg, const __bf16* __restrict__ hpre,
                           const float* __restrict__ dis, const float* __restrict__ bias,
                           __bf16* __restrict__ hout, int N) {
    long idx = (long)blockIdx.x * blockDim.x + threadIdx.x;
    if (idx >= (long)N * 128) return;
    const int i = (int)(idx >> 7);
    const int f = (int)(idx & 127);
    const float di = dis[i];
    const float t = agg[idx] + bf2f(hpre[idx]) * di * di + bias[f];
    hout[idx] = f2bf(lrelu(t));
}

// ---------------------------------------------------------------------------
// Host-side orchestration
// ---------------------------------------------------------------------------
extern "C" void kernel_launch(void* const* d_in, const int* in_sizes, int n_in,
                              void* d_out, int out_size, void* d_ws, size_t ws_size,
                              hipStream_t stream) {
    const float*     x        = (const float*)d_in[0];
    const long long* ei       = (const long long*)d_in[1];
    const float*     W_shared = (const float*)d_in[2];
    const float*     b_shared = (const float*)d_in[3];
    const float*     W_conv1  = (const float*)d_in[4];
    const float*     b_conv1  = (const float*)d_in[5];
    const float*     W_conv2  = (const float*)d_in[6];
    const float*     b_conv2  = (const float*)d_in[7];
    const float*     W_mlp1a  = (const float*)d_in[8];
    const float*     b_mlp1a  = (const float*)d_in[9];
    const float*     W_mlp1b  = (const float*)d_in[10];
    const float*     b_mlp1b  = (const float*)d_in[11];
    const float*     W_mlp2a  = (const float*)d_in[12];
    const float*     b_mlp2a  = (const float*)d_in[13];
    const float*     W_mlp2b  = (const float*)d_in[14];
    const float*     b_mlp2b  = (const float*)d_in[15];

    const int  IN = 256, HID = 128, OUT = 64;
    const int  N = in_sizes[0] / IN;        // 100000 (multiple of 16)
    const long E = (long)in_sizes[1] / 2;   // 1600000

    // Workspace carve-out (256B aligned slices)
    char* ws = (char*)d_ws;
    size_t off = 0;
    auto alloc = [&](size_t bytes) -> void* {
        void* p = ws + off;
        off = (off + bytes + 255) & ~(size_t)255;
        return p;
    };
    __bf16* Xbf  = (__bf16*)alloc((size_t)N * IN  * 2);
    __bf16* hbf  = (__bf16*)alloc((size_t)N * HID * 2);
    __bf16* h1bf = (__bf16*)alloc((size_t)N * HID * 2);
    __bf16* h2bf = (__bf16*)alloc((size_t)N * HID * 2);
    __bf16* hpre = (__bf16*)alloc((size_t)N * HID * 2);   // also reused as MLP hidden z
    float*  agg  = (float*) alloc((size_t)N * HID * 4);
    float*  deg  = (float*) alloc((size_t)N * 4);
    float*  dis  = (float*) alloc((size_t)N * 4);
    __bf16* WtS  = (__bf16*)alloc((size_t)IN  * HID * 2);
    __bf16* Wt1  = (__bf16*)alloc((size_t)HID * HID * 2);
    __bf16* Wt2  = (__bf16*)alloc((size_t)HID * HID * 2);
    __bf16* Wm1a = (__bf16*)alloc((size_t)HID * HID * 2);
    __bf16* Wm1b = (__bf16*)alloc((size_t)HID * OUT * 2);
    __bf16* Wm2a = (__bf16*)alloc((size_t)HID * HID * 2);
    __bf16* Wm2b = (__bf16*)alloc((size_t)HID * OUT * 2);

    // --- prep: bf16 conversions / weight transposes -------------------------
    {
        long nx = (long)N * IN;
        k_f32_to_bf16<<<(unsigned)((nx + 255) / 256), 256, 0, stream>>>(x, Xbf, nx);
        k_transpose_bf16<<<(IN * HID + 255) / 256, 256, 0, stream>>>(W_shared, WtS, IN, HID);
        k_transpose_bf16<<<(HID * HID + 255) / 256, 256, 0, stream>>>(W_conv1, Wt1, HID, HID);
        k_transpose_bf16<<<(HID * HID + 255) / 256, 256, 0, stream>>>(W_conv2, Wt2, HID, HID);
        k_transpose_bf16<<<(HID * HID + 255) / 256, 256, 0, stream>>>(W_mlp1a, Wm1a, HID, HID);
        k_transpose_bf16<<<(HID * OUT + 255) / 256, 256, 0, stream>>>(W_mlp1b, Wm1b, HID, OUT);
        k_transpose_bf16<<<(HID * HID + 255) / 256, 256, 0, stream>>>(W_mlp2a, Wm2a, HID, HID);
        k_transpose_bf16<<<(HID * OUT + 255) / 256, 256, 0, stream>>>(W_mlp2b, Wm2b, HID, OUT);
    }

    // --- degree normalization ----------------------------------------------
    k_fill_f32<<<(unsigned)(((long)N + 255) / 256), 256, 0, stream>>>(deg, 1.0f, N);
    k_deg_count<<<(unsigned)((E + 255) / 256), 256, 0, stream>>>(ei, deg, E);
    k_rsqrt<<<(N + 255) / 256, 256, 0, stream>>>(deg, dis, N);

    // --- one GCN conv layer -------------------------------------------------
    const unsigned gemmBlocksX = (unsigned)((N / 16 + 7) / 8);
    auto conv = [&](const __bf16* inA, int K, const __bf16* Wt,
                    const float* bias, __bf16* outH) {
        dim3 g(gemmBlocksX, HID / 64);
        k_gemm_wmma<<<g, 256, 0, stream>>>(inA, Wt, nullptr, hpre, N, K, HID, 0);
        long na = (long)N * HID;
        k_fill_f32<<<(unsigned)((na + 255) / 256), 256, 0, stream>>>(agg, 0.f, na);
        k_scatter<<<(unsigned)((E + 7) / 8), 256, 0, stream>>>(ei, hpre, dis, agg, E);
        k_finalize<<<(unsigned)((na + 255) / 256), 256, 0, stream>>>(agg, hpre, dis, bias, outH, N);
    };

    conv(Xbf, IN,  WtS, b_shared, hbf);
    conv(hbf, HID, Wt1, b_conv1,  h1bf);
    conv(hbf, HID, Wt2, b_conv2,  h2bf);

    // --- MLP heads ----------------------------------------------------------
    __bf16* zbf = hpre;  // reuse (convs done with hpre)
    dim3 gHid(gemmBlocksX, HID / 64);
    dim3 gOut(gemmBlocksX, OUT / 64);
    float* out1 = (float*)d_out;
    float* out2 = out1 + (size_t)N * OUT;

    k_gemm_wmma<<<gHid, 256, 0, stream>>>(h1bf, Wm1a, b_mlp1a, zbf,  N, HID, HID, 1);
    k_gemm_wmma<<<gOut, 256, 0, stream>>>(zbf,  Wm1b, b_mlp1b, out1, N, HID, OUT, 2);
    k_gemm_wmma<<<gHid, 256, 0, stream>>>(h2bf, Wm2a, b_mlp2a, zbf,  N, HID, HID, 1);
    k_gemm_wmma<<<gOut, 256, 0, stream>>>(zbf,  Wm2b, b_mlp2b, out2, N, HID, OUT, 2);
}
